// GraphAttentionSeg_8650064134304
// MI455X (gfx1250) — compile-verified, matrix operand order checked by reading, not run on hardware
//
#include <hip/hip_runtime.h>

// ---------------------------------------------------------------------------
// MI455X / gfx1250 implementation (wave32, WMMA bf16 16x16x32, f32 accum).
// Compile-only environment: B-fragment layout assumed to mirror the ISA
// A-fragment layout applied to B^T (CDNA5 ISA 7.12.2).
// Workspace requirement: (1024 + 2*M*C) floats ~= 16.8 MB.
// ---------------------------------------------------------------------------

#define Mpts 32768
#define Cch  64
#define Knb  16

typedef __attribute__((ext_vector_type(16))) __bf16 v16bf;
typedef __attribute__((ext_vector_type(8)))  __bf16 v8bf;
typedef __attribute__((ext_vector_type(8)))  float  v8f;

union FragAB { v16bf v; v8bf h[2]; };

__device__ __forceinline__ __bf16 bfc(float f)  { return (__bf16)f; }
__device__ __forceinline__ float  lrelu(float v){ return v > 0.f ? v : 0.1f * v; }

// ---- workspace layout (float offsets) -------------------------------------
enum : int {
  WS_SUM_T = 0,   WS_SQ_T = 3,   WS_SC_T = 8,   WS_BI_T = 11,
  WS_SUM_1 = 16,  WS_SQ_1 = 80,  WS_SC_1 = 144, WS_BI_1 = 208,
  WS_SUM_2 = 272, WS_SQ_2 = 336, WS_SC_2 = 400, WS_BI_2 = 464,
  WS_SUM_Y = 528, WS_SQ_Y = 592, WS_SC_Y = 656, WS_BI_Y = 720,
  WS_SUM_U = 784, WS_SQ_U = 848, WS_SC_U = 912, WS_BI_U = 976,
  WS_Y     = 1024,
  WS_U     = WS_Y + Mpts * Cch,
};

// ---------------------------------------------------------------------------
__global__ void k_zero(float* ws) {
  int e = blockIdx.x * 256 + threadIdx.x;
  if (e < 1024) ws[e] = 0.f;
}

// stats of t = n_r @ lp_w1 + lp_b1 over all (M,K) pairs, 3 channels
__global__ void k_lp_stats(const float* __restrict__ p, const int* __restrict__ idx,
                           const float* __restrict__ w1, const float* __restrict__ b1,
                           float* __restrict__ ws) {
  __shared__ float s[3], q[3];
  if (threadIdx.x < 3) { s[threadIdx.x] = 0.f; q[threadIdx.x] = 0.f; }
  __syncthreads();
  int e = blockIdx.x * 256 + threadIdx.x;
  if (e < Mpts * Knb) {
    int i  = e >> 4;
    int nb = idx[e];
    float n0 = p[nb*3+0] - p[i*3+0];
    float n1 = p[nb*3+1] - p[i*3+1];
    float n2 = p[nb*3+2] - p[i*3+2];
    #pragma unroll
    for (int c = 0; c < 3; ++c) {
      float t = n0 * w1[c] + n1 * w1[3+c] + n2 * w1[6+c] + b1[c];
      atomicAdd(&s[c], t);
      atomicAdd(&q[c], t * t);
    }
  }
  __syncthreads();
  if (threadIdx.x < 3) {
    atomicAdd(&ws[WS_SUM_T + threadIdx.x], s[threadIdx.x]);
    atomicAdd(&ws[WS_SQ_T  + threadIdx.x], q[threadIdx.x]);
  }
}

// (sum,sumsq,gamma,beta) -> per-channel (scale,bias) for y' = y*scale + bias
__global__ void k_finalize(const float* __restrict__ sum, const float* __restrict__ sq,
                           const float* __restrict__ g, const float* __restrict__ be,
                           float* __restrict__ scale, float* __restrict__ bias,
                           int n, float invN) {
  int c = threadIdx.x;
  if (c < n) {
    float mu  = sum[c] * invN;
    float var = sq[c] * invN - mu * mu;
    float sc  = g[c] * rsqrtf(var + 1e-5f);
    scale[c] = sc;
    bias[c]  = be[c] - mu * sc;
  }
}

// ---------------------------------------------------------------------------
// EdgeConv passes. One wave per point: 16-neighbor rows form the WMMA M-tile.
//  PHASE 1: z1 = feat @ W1                    -> accumulate stats1
//  PHASE 2: a1 = lrelu(bn1(z1)); z2 = a1 @ W2 -> accumulate stats2
//  PHASE 3: a2 = lrelu(bn2(z2)); y = max_K(a2 + pe); store y, stats_y
// ---------------------------------------------------------------------------
#define SM_W1T  0                       // [64][136] bf16, W1^T   (17408 B)
#define SM_W2T  17408                   // [64][72]  bf16, W2^T   ( 9216 B)
#define SM_SLDS 26624                   // float[64]
#define SM_QLDS 26880                   // float[64]
#define SM_H    27136                   // float[8][16][3]        ( 1536 B)
#define SM_WAVE 28672                   // 8 * 4352 B per-wave tile region
#define SM_EDGE_TOTAL 63488

template<int PHASE>
__launch_bounds__(256)
__global__ void k_edge(const float* __restrict__ p, const float* __restrict__ x,
                       const int* __restrict__ idx,
                       const float* __restrict__ W1, const float* __restrict__ W2,
                       const float* __restrict__ lpw1, const float* __restrict__ lpb1,
                       const float* __restrict__ lpw2, const float* __restrict__ lpb2,
                       float* __restrict__ ws) {
  __shared__ __align__(16) unsigned char smem[SM_EDGE_TOTAL];
  __bf16* w1t  = (__bf16*)(smem + SM_W1T);        // pitch 136
  __bf16* w2t  = (__bf16*)(smem + SM_W2T);        // pitch 72
  float*  sLds = (float*) (smem + SM_SLDS);
  float*  qLds = (float*) (smem + SM_QLDS);
  float*  hLds = (float*) (smem + SM_H);

  const int tid  = threadIdx.x;
  const int wave = tid >> 5, lane = tid & 31;
  const int row  = lane & 15, kh = lane >> 4;     // C-layout: N=row, M in {kh*8..kh*8+7}
  const int point = blockIdx.x * 8 + wave;

  unsigned char* wbase = smem + SM_WAVE + wave * 4352;
  __bf16* feat  = (__bf16*)wbase;                 // [16][136] bf16 (aliased)
  __bf16* a1t   = (__bf16*)wbase;                 // [16][72]  bf16 (reuses feat region)
  float*  pairm = (float*) (wbase + 2304);        // [32] cross-half scratch

  // stage W1^T (and W2^T) as bf16
  for (int e = tid; e < 128 * 64; e += 256) {
    int k = e >> 6, c = e & 63;
    w1t[c * 136 + k] = bfc(W1[e]);
  }
  if constexpr (PHASE >= 2) {
    for (int e = tid; e < 64 * 64; e += 256) {
      int k = e >> 6, c = e & 63;
      w2t[c * 72 + k] = bfc(W2[e]);
    }
  }
  if (tid < 64) { sLds[tid] = 0.f; qLds[tid] = 0.f; }

  // PHASE 3: positional MLP hidden h = relu(bn_t(n_r @ lp_w1 + lp_b1)), 16 rows/wave
  if constexpr (PHASE == 3) {
    if (kh == 0) {
      int nb = idx[point * Knb + row];
      float n0 = p[nb*3+0] - p[point*3+0];
      float n1 = p[nb*3+1] - p[point*3+1];
      float n2 = p[nb*3+2] - p[point*3+2];
      #pragma unroll
      for (int c = 0; c < 3; ++c) {
        float t = n0 * lpw1[c] + n1 * lpw1[3+c] + n2 * lpw1[6+c] + lpb1[c];
        t = t * ws[WS_SC_T + c] + ws[WS_BI_T + c];
        hLds[(wave * 16 + row) * 3 + c] = fmaxf(t, 0.f);
      }
    }
  }
  __syncthreads();

  // build feat tile: [K=16 rows][2C=128]: cols<64 = x_i, cols>=64 = x_nb - x_i
  const float* xi = x + point * Cch;
  for (int r = 0; r < 16; ++r) {
    int nb = idx[point * Knb + r];
    const float* xn = x + nb * Cch;
    int j = lane * 4;
    #pragma unroll
    for (int t = 0; t < 4; ++t, ++j) {
      float v = (j < Cch) ? xi[j] : (xn[j - Cch] - xi[j - Cch]);
      feat[r * 136 + j] = bfc(v);
    }
  }

  // A fragments for GEMM1 (ISA 16-bit 16x32 A layout): two ds_read_b128 each
  FragAB A1[4];
  #pragma unroll
  for (int kc = 0; kc < 4; ++kc) {
    const __bf16* src = feat + row * 136 + kc * 32 + kh * 8;
    A1[kc].h[0] = *(const v8bf*)(src);
    A1[kc].h[1] = *(const v8bf*)(src + 16);
  }

  // GEMM1: z1(16x64) = feat(16x128) @ W1(128x64)
  #pragma unroll
  for (int nt = 0; nt < 4; ++nt) {
    v8f acc = {0.f,0.f,0.f,0.f,0.f,0.f,0.f,0.f};
    #pragma unroll
    for (int kc = 0; kc < 4; ++kc) {
      FragAB B;
      const __bf16* src = w1t + (nt * 16 + row) * 136 + kc * 32 + kh * 8;
      B.h[0] = *(const v8bf*)(src);
      B.h[1] = *(const v8bf*)(src + 16);
      acc = __builtin_amdgcn_wmma_f32_16x16x32_bf16(
          false, A1[kc].v, false, B.v, (short)0, acc, false, false);
    }
    int c = nt * 16 + row;
    if constexpr (PHASE == 1) {
      float s = 0.f, q = 0.f;
      #pragma unroll
      for (int r = 0; r < 8; ++r) { float v = acc[r]; s += v; q += v * v; }
      atomicAdd(&sLds[c], s);
      atomicAdd(&qLds[c], q);
    } else {
      float sc = ws[WS_SC_1 + c], bi = ws[WS_BI_1 + c];
      #pragma unroll
      for (int r = 0; r < 8; ++r)
        a1t[(r + 8 * kh) * 72 + c] = bfc(lrelu(acc[r] * sc + bi));
    }
  }

  if constexpr (PHASE >= 2) {
    // GEMM2: z2(16x64) = a1(16x64) @ W2(64x64)
    FragAB A2[2];
    #pragma unroll
    for (int kc = 0; kc < 2; ++kc) {
      const __bf16* src = a1t + row * 72 + kc * 32 + kh * 8;
      A2[kc].h[0] = *(const v8bf*)(src);
      A2[kc].h[1] = *(const v8bf*)(src + 16);
    }
    #pragma unroll
    for (int nt = 0; nt < 4; ++nt) {
      v8f acc = {0.f,0.f,0.f,0.f,0.f,0.f,0.f,0.f};
      #pragma unroll
      for (int kc = 0; kc < 2; ++kc) {
        FragAB B;
        const __bf16* src = w2t + (nt * 16 + row) * 72 + kc * 32 + kh * 8;
        B.h[0] = *(const v8bf*)(src);
        B.h[1] = *(const v8bf*)(src + 16);
        acc = __builtin_amdgcn_wmma_f32_16x16x32_bf16(
            false, A2[kc].v, false, B.v, (short)0, acc, false, false);
      }
      int c = nt * 16 + row;
      if constexpr (PHASE == 2) {
        float s = 0.f, q = 0.f;
        #pragma unroll
        for (int r = 0; r < 8; ++r) { float v = acc[r]; s += v; q += v * v; }
        atomicAdd(&sLds[c], s);
        atomicAdd(&qLds[c], q);
      } else {
        // a2 = lrelu(bn2(z2)); add pe; max over the 16 neighbor rows
        float sc  = ws[WS_SC_2 + c], bi = ws[WS_BI_2 + c];
        float w20 = lpw2[c], w21 = lpw2[64 + c], w22 = lpw2[128 + c];
        float b2  = lpb2[c];
        float mx  = -3.4e38f;
        #pragma unroll
        for (int r = 0; r < 8; ++r) {
          int mr = r + 8 * kh;
          const float* hh = hLds + (wave * 16 + mr) * 3;
          float pe = hh[0] * w20 + hh[1] * w21 + hh[2] * w22 + b2;
          float v  = lrelu(acc[r] * sc + bi) + pe;
          mx = fmaxf(mx, v);
        }
        // combine rows 0-7 (kh=0) with rows 8-15 (kh=1): same-wave DS ops are in-order
        pairm[lane] = mx;
        float ymax = fmaxf(mx, pairm[lane ^ 16]);
        if (kh == 0) {
          ws[WS_Y + point * Cch + c] = ymax;
          atomicAdd(&sLds[c], ymax);
          atomicAdd(&qLds[c], ymax * ymax);
        }
      }
    }
  }

  __syncthreads();
  if (tid < 64) {
    int so = (PHASE == 1) ? WS_SUM_1 : (PHASE == 2) ? WS_SUM_2 : WS_SUM_Y;
    int qo = (PHASE == 1) ? WS_SQ_1  : (PHASE == 2) ? WS_SQ_2  : WS_SQ_Y;
    atomicAdd(&ws[so + tid], sLds[tid]);
    atomicAdd(&ws[qo + tid], qLds[tid]);
  }
}

// ---------------------------------------------------------------------------
// Tail linear: t = relu(bn_y(y)); u = t @ lin_w + lin_b  (WMMA over 16-point
// row tiles), accumulate u stats for blk_bn2.
// ---------------------------------------------------------------------------
__launch_bounds__(256)
__global__ void k_tail(const float* __restrict__ lin_w, const float* __restrict__ lin_b,
                       float* __restrict__ ws) {
  __shared__ __align__(16) unsigned char smem[9216 + 512 + 8 * 2304];
  __bf16* wt   = (__bf16*)smem;                   // [64][72] lin_w^T
  float*  sLds = (float*)(smem + 9216);
  float*  qLds = (float*)(smem + 9472);

  const int tid  = threadIdx.x;
  const int wave = tid >> 5, lane = tid & 31;
  const int row  = lane & 15, kh = lane >> 4;
  __bf16* tt = (__bf16*)(smem + 9728 + wave * 2304);   // [16][72]

  for (int e = tid; e < 64 * 64; e += 256) {
    int k = e >> 6, c = e & 63;
    wt[c * 72 + k] = bfc(lin_w[e]);
  }
  if (tid < 64) { sLds[tid] = 0.f; qLds[tid] = 0.f; }
  __syncthreads();

  const int i0 = (blockIdx.x * 8 + wave) * 16;    // 16 points per wave
  for (int e = lane; e < 16 * 64; e += 32) {
    int r = e >> 6, c = e & 63;
    float v = ws[WS_Y + (i0 + r) * Cch + c];
    tt[r * 72 + c] = bfc(fmaxf(v * ws[WS_SC_Y + c] + ws[WS_BI_Y + c], 0.f));
  }

  FragAB A[2];
  #pragma unroll
  for (int kc = 0; kc < 2; ++kc) {
    const __bf16* src = tt + row * 72 + kc * 32 + kh * 8;
    A[kc].h[0] = *(const v8bf*)(src);
    A[kc].h[1] = *(const v8bf*)(src + 16);
  }
  #pragma unroll
  for (int nt = 0; nt < 4; ++nt) {
    v8f acc = {0.f,0.f,0.f,0.f,0.f,0.f,0.f,0.f};
    #pragma unroll
    for (int kc = 0; kc < 2; ++kc) {
      FragAB B;
      const __bf16* src = wt + (nt * 16 + row) * 72 + kc * 32 + kh * 8;
      B.h[0] = *(const v8bf*)(src);
      B.h[1] = *(const v8bf*)(src + 16);
      acc = __builtin_amdgcn_wmma_f32_16x16x32_bf16(
          false, A[kc].v, false, B.v, (short)0, acc, false, false);
    }
    int c = nt * 16 + row;
    float bb = lin_b[c];
    float s = 0.f, q = 0.f;
    #pragma unroll
    for (int r = 0; r < 8; ++r) {
      float u = acc[r] + bb;
      ws[WS_U + (i0 + r + 8 * kh) * Cch + c] = u;
      s += u; q += u * u;
    }
    atomicAdd(&sLds[c], s);
    atomicAdd(&qLds[c], q);
  }
  __syncthreads();
  if (tid < 64) {
    atomicAdd(&ws[WS_SUM_U + tid], sLds[tid]);
    atomicAdd(&ws[WS_SQ_U  + tid], qLds[tid]);
  }
}

// out = relu(bn_u(u) + x)
__global__ void k_final(const float* __restrict__ x, const float* __restrict__ ws,
                        float* __restrict__ out) {
  int e = blockIdx.x * 256 + threadIdx.x;
  if (e < Mpts * Cch) {
    int c = e & 63;
    float v = ws[WS_U + e] * ws[WS_SC_U + c] + ws[WS_BI_U + c] + x[e];
    out[e] = fmaxf(v, 0.f);
  }
}

// ---------------------------------------------------------------------------
extern "C" void kernel_launch(void* const* d_in, const int* in_sizes, int n_in,
                              void* d_out, int out_size, void* d_ws, size_t ws_size,
                              hipStream_t stream) {
  (void)in_sizes; (void)n_in; (void)out_size; (void)ws_size;
  const float* p       = (const float*)d_in[0];
  const float* x       = (const float*)d_in[1];
  const int*   idx     = (const int*)  d_in[2];
  const float* lp_w1   = (const float*)d_in[3];
  const float* lp_b1   = (const float*)d_in[4];
  const float* lp_g    = (const float*)d_in[5];
  const float* lp_be   = (const float*)d_in[6];
  const float* lp_w2   = (const float*)d_in[7];
  const float* lp_b2   = (const float*)d_in[8];
  const float* conv1_w = (const float*)d_in[9];
  const float* bn1_g   = (const float*)d_in[10];
  const float* bn1_b   = (const float*)d_in[11];
  const float* conv2_w = (const float*)d_in[12];
  const float* bn2_g   = (const float*)d_in[13];
  const float* bn2_b   = (const float*)d_in[14];
  const float* blk1_g  = (const float*)d_in[15];
  const float* blk1_b  = (const float*)d_in[16];
  const float* lin_w   = (const float*)d_in[17];
  const float* lin_b   = (const float*)d_in[18];
  const float* blk2_g  = (const float*)d_in[19];
  const float* blk2_b  = (const float*)d_in[20];
  float* ws  = (float*)d_ws;
  float* out = (float*)d_out;

  const float invMK = 1.f / (float)(Mpts * Knb);
  const float invM  = 1.f / (float)Mpts;

  k_zero<<<4, 256, 0, stream>>>(ws);
  k_lp_stats<<<(Mpts * Knb) / 256, 256, 0, stream>>>(p, idx, lp_w1, lp_b1, ws);
  k_finalize<<<1, 64, 0, stream>>>(ws + WS_SUM_T, ws + WS_SQ_T, lp_g, lp_be,
                                   ws + WS_SC_T, ws + WS_BI_T, 3, invMK);
  k_edge<1><<<Mpts / 8, 256, 0, stream>>>(p, x, idx, conv1_w, conv2_w,
                                          lp_w1, lp_b1, lp_w2, lp_b2, ws);
  k_finalize<<<1, 64, 0, stream>>>(ws + WS_SUM_1, ws + WS_SQ_1, bn1_g, bn1_b,
                                   ws + WS_SC_1, ws + WS_BI_1, 64, invMK);
  k_edge<2><<<Mpts / 8, 256, 0, stream>>>(p, x, idx, conv1_w, conv2_w,
                                          lp_w1, lp_b1, lp_w2, lp_b2, ws);
  k_finalize<<<1, 64, 0, stream>>>(ws + WS_SUM_2, ws + WS_SQ_2, bn2_g, bn2_b,
                                   ws + WS_SC_2, ws + WS_BI_2, 64, invMK);
  k_edge<3><<<Mpts / 8, 256, 0, stream>>>(p, x, idx, conv1_w, conv2_w,
                                          lp_w1, lp_b1, lp_w2, lp_b2, ws);
  k_finalize<<<1, 64, 0, stream>>>(ws + WS_SUM_Y, ws + WS_SQ_Y, blk1_g, blk1_b,
                                   ws + WS_SC_Y, ws + WS_BI_Y, 64, invM);
  k_tail<<<Mpts / 128, 256, 0, stream>>>(lin_w, lin_b, ws);
  k_finalize<<<1, 64, 0, stream>>>(ws + WS_SUM_U, ws + WS_SQ_U, blk2_g, blk2_b,
                                   ws + WS_SC_U, ws + WS_BI_U, 64, invM);
  k_final<<<(Mpts * Cch) / 256, 256, 0, stream>>>(x, ws, out);
}
